// SelfAttention_v2_9028021256254
// MI455X (gfx1250) — matrix operand order repaired and checked by reading, not compiled, not run on hardware
//
#include <hip/hip_runtime.h>

// CDNA5 / gfx1250 fused self-attention, transposed-score formulation:
//   proj_qkv  : q,k,v projections (fp32 accum) -> bf16; q pre-scaled by
//               (1/sqrt(64))*log2(e) so softmax runs in base 2; v transposed
//   attn_fused: flash-style online softmax; S^T = K Q^T puts queries on lanes
//               (in-register reductions + v_permlanex16, no LDS transpose),
//               K-tile operands double-buffered so loads overlap WMMAs.
typedef __attribute__((ext_vector_type(16))) __bf16 v16bf;
typedef __attribute__((ext_vector_type(2)))  __bf16 v2bf;
typedef __attribute__((ext_vector_type(8)))  float  v8f;

#define N_TOK   8192
#define D_IN    768
#define D_OUT   64
#define ATT_WAVES 4

union V16 {
  v16bf    v;
  uint4    q[2];
  unsigned u[8];
};

// native f32 -> bf16 (RNE); backend selects v_cvt_*bf16* ops
__device__ __forceinline__ unsigned short f2bf(float f) {
  __bf16 b = (__bf16)f;
  return __builtin_bit_cast(unsigned short, b);
}
__device__ __forceinline__ unsigned pack_bf16(float lo, float hi) {
  v2bf p;
  p[0] = (__bf16)lo;
  p[1] = (__bf16)hi;
  return __builtin_bit_cast(unsigned, p);
}

// exchange with the partner lane in the other 16-lane half (lane ^ 16)
__device__ __forceinline__ unsigned xhalf_u(unsigned v) {
#if __has_builtin(__builtin_amdgcn_permlanex16)
  return (unsigned)__builtin_amdgcn_permlanex16((int)v, (int)v, 0, 0, false, false);
#else
  return (unsigned)__shfl_xor((int)v, 16);
#endif
}
__device__ __forceinline__ float xhalf_f(float v) {
  return __builtin_bit_cast(float, xhalf_u(__builtin_bit_cast(unsigned, v)));
}

__device__ __forceinline__ float fast_exp2(float x) {
#if __has_builtin(__builtin_amdgcn_exp2f)
  return __builtin_amdgcn_exp2f(x);          // v_exp_f32
#else
  return __expf(x * 0.69314718056f);
#endif
}

// ---------------- projections: q = X Wq * c, k = X Wk, v = X Wv -------------
__global__ __launch_bounds__(256) void proj_qkv(
    const float* __restrict__ X,  const float* __restrict__ Wq,
    const float* __restrict__ Wk, const float* __restrict__ Wv,
    unsigned short* __restrict__ qbf, unsigned short* __restrict__ kbf,
    unsigned short* __restrict__ vtbf)
{
  __shared__ float xs[16 * D_IN];            // 48 KB tile of X
  const int t = threadIdx.x;
  const int rowbase = blockIdx.x * 16;
  for (int i = t; i < 16 * D_IN; i += 256)
    xs[i] = X[(size_t)rowbase * D_IN + i];
  __syncthreads();

  const int col = t & 63;                    // output column 0..63
  const int r0  = (t >> 6) * 4;              // 4 rows per thread
  float aq[4] = {0,0,0,0}, ak[4] = {0,0,0,0}, av[4] = {0,0,0,0};
  for (int kk = 0; kk < D_IN; ++kk) {
    const float wq = Wq[kk * D_OUT + col];
    const float wk = Wk[kk * D_OUT + col];
    const float wv = Wv[kk * D_OUT + col];
    #pragma unroll
    for (int j = 0; j < 4; ++j) {
      const float x = xs[(r0 + j) * D_IN + kk];
      aq[j] = fmaf(x, wq, aq[j]);
      ak[j] = fmaf(x, wk, ak[j]);
      av[j] = fmaf(x, wv, av[j]);
    }
  }
  // fold 1/sqrt(64) and log2(e) into q so attention softmax uses exp2
  const float QSCALE = 0.125f * 1.4426950408889634f;
  #pragma unroll
  for (int j = 0; j < 4; ++j) {
    const int row = rowbase + r0 + j;
    qbf[row * D_OUT + col]          = f2bf(aq[j] * QSCALE);
    kbf[row * D_OUT + col]          = f2bf(ak[j]);
    vtbf[(size_t)col * N_TOK + row] = f2bf(av[j]);           // v transposed
  }
}

// Load the four 16x32-feat K-tile A-operands for keys [j0, j0+32).
__device__ __forceinline__ void load_ka(V16 ka[4], const unsigned short* kbf,
                                        int j0, int l16, int g) {
  const unsigned short* kp0 = kbf + (j0 + l16) * D_OUT + g * 8;
  const unsigned short* kp1 = kp0 + 16 * D_OUT;
  ka[0].q[0] = *(const uint4*)(kp0);      ka[0].q[1] = *(const uint4*)(kp0 + 16);
  ka[1].q[0] = *(const uint4*)(kp0 + 32); ka[1].q[1] = *(const uint4*)(kp0 + 48);
  ka[2].q[0] = *(const uint4*)(kp1);      ka[2].q[1] = *(const uint4*)(kp1 + 16);
  ka[3].q[0] = *(const uint4*)(kp1 + 32); ka[3].q[1] = *(const uint4*)(kp1 + 48);
}

// Process one 32-key chunk: scores, online softmax (base 2), O^T += V^T P^T.
__device__ __forceinline__ void attn_chunk(
    const V16 ka[4], const V16& qb0, const V16& qb1,
    const unsigned short* __restrict__ vtbf, int j0, int l16, int g,
    v8f o[4], float& m_run, float& l_run)
{
  const v8f zacc = {0.f,0.f,0.f,0.f,0.f,0.f,0.f,0.f};

  // V^T A-operands first so they are in flight across the score WMMAs
  V16 va[4];
  #pragma unroll
  for (int c = 0; c < 4; ++c) {
    const unsigned short* vp = vtbf + (size_t)(c * 16 + l16) * N_TOK + j0 + g * 16;
    va[c].q[0] = *(const uint4*)(vp);
    va[c].q[1] = *(const uint4*)(vp + 8);
  }

  // S^T tiles: t0 = keys j0 + (r + 8g), t1 = keys j0+16 + (r + 8g); lane = query
  v8f t0 = __builtin_amdgcn_wmma_f32_16x16x32_bf16(false, ka[0].v, false, qb0.v, (short)0, zacc, false, false);
  t0     = __builtin_amdgcn_wmma_f32_16x16x32_bf16(false, ka[1].v, false, qb1.v, (short)0, t0,   false, false);
  v8f t1 = __builtin_amdgcn_wmma_f32_16x16x32_bf16(false, ka[2].v, false, qb0.v, (short)0, zacc, false, false);
  t1     = __builtin_amdgcn_wmma_f32_16x16x32_bf16(false, ka[3].v, false, qb1.v, (short)0, t1,   false, false);

  // ---- online softmax (base-2), all in-register ----
  float m_own = fmaxf(t0[0], t1[0]);
  #pragma unroll
  for (int r = 1; r < 8; ++r) m_own = fmaxf(m_own, fmaxf(t0[r], t1[r]));
  const float m_chunk = fmaxf(m_own, xhalf_f(m_own));   // combine halves
  const float mnew    = fmaxf(m_run, m_chunk);
  const float alpha   = fast_exp2(m_run - mnew);
  m_run = mnew;

  float e0[8], e1[8], s_own = 0.f;
  #pragma unroll
  for (int r = 0; r < 8; ++r) {
    e0[r] = fast_exp2(t0[r] - mnew);
    e1[r] = fast_exp2(t1[r] - mnew);
    s_own += e0[r] + e1[r];
  }
  l_run = l_run * alpha + s_own + xhalf_f(s_own);

  // ---- assemble P^T as B operand (32 keys x 16 queries) in registers ----
  // B vgpr v of lane (q,g) packs chunk-keys (16g+2v, 16g+2v+1); fetch the
  // missing half-rows from the partner lane via permlanex16.
  unsigned A0[4], A1[4];
  #pragma unroll
  for (int v = 0; v < 4; ++v) {
    A0[v] = pack_bf16(e0[2*v], e0[2*v+1]);
    A1[v] = pack_bf16(e1[2*v], e1[2*v+1]);
  }
  V16 pb;
  #pragma unroll
  for (int v = 0; v < 4; ++v) {
    const unsigned x0 = xhalf_u(A0[v]);    // partner's t0 exps
    const unsigned x1 = xhalf_u(A1[v]);    // partner's t1 exps
    pb.u[v]     = g ? x1    : A0[v];       // keys 16g+0..7
    pb.u[v + 4] = g ? A1[v] : x0;          // keys 16g+8..15
  }

  // rescale accumulators (alpha uniform per lane/query), then accumulate
  #pragma unroll
  for (int c = 0; c < 4; ++c) o[c] *= alpha;
  #pragma unroll
  for (int c = 0; c < 4; ++c)
    o[c] = __builtin_amdgcn_wmma_f32_16x16x32_bf16(false, va[c].v, false, pb.v, (short)0, o[c], false, false);
}

// ---------------- fused attention: softmax(q k^T / 8) v ---------------------
// One wave owns 16 queries; keys processed 32 at a time, K operands
// double-buffered (ping-pong, unroll by 2) to overlap loads with WMMAs.
__global__ __launch_bounds__(32 * ATT_WAVES) void attn_fused(
    const unsigned short* __restrict__ qbf,
    const unsigned short* __restrict__ kbf,
    const unsigned short* __restrict__ vtbf,
    float* __restrict__ out)
{
  const int lane  = threadIdx.x & 31;
  const int l16   = lane & 15;
  const int g     = lane >> 4;               // 16-lane half group
  const int wave  = threadIdx.x >> 5;
  const int qbase = (blockIdx.x * ATT_WAVES + wave) * 16;

  // B operand = Q^T (32 feat x 16 queries): lane = query column l16,
  // element e -> feature 32*step + 8*g + e (e<8), 32*step + 16 + 8*g + (e-8) (e>=8).
  V16 qb0, qb1;
  {
    const unsigned short* qp = qbf + (qbase + l16) * D_OUT + g * 8;
    qb0.q[0] = *(const uint4*)(qp);      qb0.q[1] = *(const uint4*)(qp + 16);
    qb1.q[0] = *(const uint4*)(qp + 32); qb1.q[1] = *(const uint4*)(qp + 48);
  }

  v8f o[4];                                  // O^T tiles: vgpr = vcol, lane = query
  #pragma unroll
  for (int c = 0; c < 4; ++c) o[c] = (v8f){0.f,0.f,0.f,0.f,0.f,0.f,0.f,0.f};
  float m_run = -INFINITY, l_run = 0.f;

  V16 kaA[4], kaB[4];
  load_ka(kaA, kbf, 0, l16, g);

  for (int j0 = 0; j0 < N_TOK; j0 += 64) {
    if (j0 + 64 < N_TOK) {                   // global_prefetch_b8 next pair
      __builtin_prefetch(kbf + (j0 + 64 + lane) * D_OUT, 0, 0);
      __builtin_prefetch(vtbf + (size_t)lane * N_TOK + j0 + 64, 0, 0);
    }
    load_ka(kaB, kbf, j0 + 32, l16, g);      // in flight during chunk j0
    attn_chunk(kaA, qb0, qb1, vtbf, j0, l16, g, o, m_run, l_run);
    if (j0 + 64 < N_TOK)
      load_ka(kaA, kbf, j0 + 64, l16, g);    // in flight during chunk j0+32
    attn_chunk(kaB, qb0, qb1, vtbf, j0 + 32, l16, g, o, m_run, l_run);
  }

  // normalize and store: lane (q,g) holds vcols c*16 + 8g + r for query l16
  const float inv = __frcp_rn(l_run);
  float* op = out + (qbase + l16) * D_OUT + g * 8;
  #pragma unroll
  for (int c = 0; c < 4; ++c) {
    float4 lo, hi;
    lo.x = o[c][0] * inv; lo.y = o[c][1] * inv;
    lo.z = o[c][2] * inv; lo.w = o[c][3] * inv;
    hi.x = o[c][4] * inv; hi.y = o[c][5] * inv;
    hi.z = o[c][6] * inv; hi.w = o[c][7] * inv;
    *(float4*)(op + c * 16)     = lo;
    *(float4*)(op + c * 16 + 4) = hi;
  }
}

extern "C" void kernel_launch(void* const* d_in, const int* in_sizes, int n_in,
                              void* d_out, int out_size, void* d_ws, size_t ws_size,
                              hipStream_t stream) {
  (void)in_sizes; (void)n_in; (void)out_size; (void)ws_size;
  const float* X  = (const float*)d_in[0];
  const float* Wq = (const float*)d_in[1];
  const float* Wk = (const float*)d_in[2];
  const float* Wv = (const float*)d_in[3];
  float* out = (float*)d_out;

  // workspace: bf16 q[N][64] (pre-scaled), k[N][64], v^T[64][N]  (3 MB total)
  unsigned short* ws   = (unsigned short*)d_ws;
  unsigned short* qbf  = ws;
  unsigned short* kbf  = ws + (size_t)N_TOK * D_OUT;
  unsigned short* vtbf = ws + 2 * (size_t)N_TOK * D_OUT;

  proj_qkv<<<N_TOK / 16, 256, 0, stream>>>(X, Wq, Wk, Wv, qbf, kbf, vtbf);
  attn_fused<<<N_TOK / (16 * ATT_WAVES), 32 * ATT_WAVES, 0, stream>>>(qbf, kbf, vtbf, out);
}
